// Attention_6760278524268
// MI455X (gfx1250) — matrix operand order repaired and checked by reading, not compiled, not run on hardware
//
#include <hip/hip_runtime.h>

typedef unsigned short u16;
typedef unsigned int u32;
typedef __attribute__((ext_vector_type(16))) __bf16 v16bf;
typedef __attribute__((ext_vector_type(8))) float v8f;

constexpr int B = 2, S = 2048, H = 2048;
constexpr int NQ = 32, NKV = 8, G = 4, D = 64;
constexpr int BS = B * S;    // 4096
constexpr int HKV = H / G;   // 512

// ---------------- workspace layout (bytes) ----------------
constexpr size_t OFF_HSB = 0;                               // bf16 hidden [BS,H]
constexpr size_t OFF_WQT = OFF_HSB + (size_t)BS * H * 2;    // bf16 wq^T [H,H]
constexpr size_t OFF_WKT = OFF_WQT + (size_t)H * H * 2;     // bf16 wk^T [HKV,H]
constexpr size_t OFF_WVT = OFF_WKT + (size_t)HKV * H * 2;   // bf16 wv^T [HKV,H]
constexpr size_t OFF_WOT = OFF_WVT + (size_t)HKV * H * 2;   // bf16 wo^T [H,H]
constexpr size_t OFF_QF  = OFF_WOT + (size_t)H * H * 2;     // f32 Q pre-rope [BS,H]
constexpr size_t OFF_KF  = OFF_QF + (size_t)BS * H * 4;     // f32 K pre-rope [BS,HKV]
constexpr size_t OFF_VF  = OFF_KF + (size_t)BS * HKV * 4;   // f32 V [BS,HKV]
constexpr size_t OFF_QB  = OFF_VF + (size_t)BS * HKV * 4;   // bf16 Q [B,NQ,S,D]
constexpr size_t OFF_KB  = OFF_QB + (size_t)BS * H * 2;     // bf16 K [B,NKV,S,D]
constexpr size_t OFF_VT  = OFF_KB + (size_t)BS * HKV * 2;   // bf16 V^T [B,NKV,D,S]
constexpr size_t OFF_AB  = OFF_VT + (size_t)BS * HKV * 2;   // bf16 attn out [BS,H]
// total ~124 MiB

// ---------------- helpers ----------------
static __device__ __forceinline__ u16 f2bf(float f) {
  u32 u = __builtin_bit_cast(u32, f);
  u32 r = u + 0x7FFFu + ((u >> 16) & 1u);  // round-to-nearest-even
  return (u16)(r >> 16);
}
struct Pack32 { uint4 lo, hi; };
static __device__ __forceinline__ v16bf mkfrag(uint4 lo, uint4 hi) {
  Pack32 p{lo, hi};
  return __builtin_bit_cast(v16bf, p);
}
static __device__ __forceinline__ uint4 ld4(const u16* p) {
  return *reinterpret_cast<const uint4*>(p);
}
static __device__ __forceinline__ v8f vzero() {
  v8f z;
#pragma unroll
  for (int i = 0; i < 8; ++i) z[i] = 0.0f;
  return z;
}
static __device__ __forceinline__ v8f wmma_bf16(v16bf a, v16bf b, v8f c) {
  return __builtin_amdgcn_wmma_f32_16x16x32_bf16(false, a, false, b, (short)0, c,
                                                 false, false);
}

// ---------------- elementwise conversion kernels ----------------
__global__ void k_cvt(const float* __restrict__ x, u16* __restrict__ y, int n) {
  int i = blockIdx.x * blockDim.x + threadIdx.x;
  if (i < n) y[i] = f2bf(x[i]);
}

// w [K,N] f32 row-major -> wt [N,K] bf16 row-major
__global__ void k_wtrans(const float* __restrict__ w, u16* __restrict__ wt, int K, int N) {
  int i = blockIdx.x * blockDim.x + threadIdx.x;
  if (i >= K * N) return;
  int k = i / N, n = i - k * N;
  wt[(size_t)n * K + k] = f2bf(w[i]);
}

// ---------------- bf16 WMMA GEMM: C[M,N] f32 = A[M,K]bf16 * Bt[N,K]bf16^T ----------------
__global__ __launch_bounds__(256) void k_gemm(const u16* __restrict__ A,
                                              const u16* __restrict__ Bt,
                                              float* __restrict__ C,
                                              int M, int N, int K) {
  const int lane = threadIdx.x & 31;
  const int wave = threadIdx.x >> 5;
  const int l15 = lane & 15, hf = lane >> 4;
  const int m0 = blockIdx.y * 128 + (wave >> 1) * 32;
  const int n0 = blockIdx.x * 128 + (wave & 1) * 64;
  v8f acc[2][4];
#pragma unroll
  for (int i = 0; i < 2; ++i)
#pragma unroll
    for (int t = 0; t < 4; ++t) acc[i][t] = vzero();

  // A-fragment per-lane base: row m0+l15, K-chunks at hf*8 and 16+hf*8
  const u16* a0p = A + (size_t)(m0 + l15) * K + hf * 8;
  const u16* a1p = a0p + (size_t)16 * K;
  // B-fragment per-lane base: row n0+t*16+l15, K-chunks at hf*8 and 16+hf*8
  const u16* bp[4];
#pragma unroll
  for (int t = 0; t < 4; ++t) bp[t] = Bt + (size_t)(n0 + t * 16 + l15) * K + hf * 8;

  for (int k0 = 0; k0 < K; k0 += 32) {
    // issue ALL fragment loads first so they land in one clause and the
    // scheduler can retire waits progressively against the 8 WMMAs below
    uint4 alo0 = ld4(a0p + k0), ahi0 = ld4(a0p + k0 + 16);
    uint4 alo1 = ld4(a1p + k0), ahi1 = ld4(a1p + k0 + 16);
    uint4 blo[4], bhi[4];
#pragma unroll
    for (int t = 0; t < 4; ++t) {
      blo[t] = ld4(bp[t] + k0);
      bhi[t] = ld4(bp[t] + k0 + 16);
    }
    if (k0 + 64 < K) {  // prefetch two K-steps ahead, WGP scope (high locality)
      __builtin_prefetch(a0p + k0 + 64, 0, 3);
      __builtin_prefetch(a1p + k0 + 64, 0, 3);
      __builtin_prefetch(bp[0] + k0 + 64, 0, 3);
      __builtin_prefetch(bp[2] + k0 + 64, 0, 3);
    }
    v16bf a0 = mkfrag(alo0, ahi0);
    v16bf a1 = mkfrag(alo1, ahi1);
#pragma unroll
    for (int t = 0; t < 4; ++t) {
      v16bf b = mkfrag(blo[t], bhi[t]);
      acc[0][t] = wmma_bf16(a0, b, acc[0][t]);
      acc[1][t] = wmma_bf16(a1, b, acc[1][t]);
    }
  }
#pragma unroll
  for (int i = 0; i < 2; ++i)
#pragma unroll
    for (int t = 0; t < 4; ++t)
#pragma unroll
      for (int r = 0; r < 8; ++r)
        C[(size_t)(m0 + i * 16 + r + 8 * hf) * N + (n0 + t * 16 + l15)] = acc[i][t][r];
}

// ---------------- RoPE + repack [B,S,NH,D]f32 -> [B,NH,S,D]bf16 ----------------
__global__ void k_rope(const float* __restrict__ x, const int* __restrict__ pos,
                       u16* __restrict__ out, int NH) {
  int idx = blockIdx.x * blockDim.x + threadIdx.x;
  int total = B * NH * S * (D / 2);
  if (idx >= total) return;
  int j = idx & (D / 2 - 1);          // 0..31 pair index
  int rest = idx >> 5;
  int s = rest & (S - 1);
  rest >>= 11;
  int h = rest % NH;
  int b = rest / NH;
  int p = pos[b * S + s];
  // freq = theta^(-2j/D) ; ln(10000)/64 = 0.14391156...
  float freq = __expf(-(float)(2 * j) * 0.14391156516f);
  float ang = (float)p * freq;
  float sn, cs;
  __sincosf(ang, &sn, &cs);
  size_t ib = (((size_t)(b * S + s)) * NH + h) * D + 2 * j;
  float xr = x[ib], xi = x[ib + 1];
  float orr = xr * cs - xi * sn;
  float oi = xr * sn + xi * cs;
  size_t ob = (((size_t)(b * NH + h)) * S + s) * D + 2 * j;
  u32 pk = (u32)f2bf(orr) | ((u32)f2bf(oi) << 16);
  *reinterpret_cast<u32*>(out + ob) = pk;
}

// ---------------- V: [B,S,NKV,D]f32 -> [B,NKV,D,S]bf16 (transposed for B-frags) -------
__global__ void k_vtrans(const float* __restrict__ vf, u16* __restrict__ vt) {
  int idx = blockIdx.x * blockDim.x + threadIdx.x;
  int total = B * NKV * S * D;
  if (idx >= total) return;
  int d = idx & (D - 1);
  int s = (idx >> 6) & (S - 1);
  int h = (idx >> 17) & (NKV - 1);
  int b = idx >> 20;
  vt[(((size_t)(b * NKV + h)) * D + d) * S + s] =
      f2bf(vf[(((size_t)(b * S + s)) * NKV + h) * D + d]);
}

// ---------------- flash attention (causal, GQA) ----------------
// grid (S/64, NQ, B), block 128 (4 waves); each wave owns 16 query rows.
__global__ __launch_bounds__(128) void k_flash(const u16* __restrict__ qb,
                                               const u16* __restrict__ kbuf,
                                               const u16* __restrict__ vtb,
                                               const int* __restrict__ amask,
                                               u16* __restrict__ ab) {
  __shared__ u16 lds[4][16 * 64];  // per-wave P tile (bf16)
  const int lane = threadIdx.x & 31;
  const int wave = threadIdx.x >> 5;
  const int l15 = lane & 15, hf = lane >> 4;
  const int b = blockIdx.z, qh = blockIdx.y, kvh = qh / G;
  const int q0 = blockIdx.x * 64 + wave * 16;
  const u16* Q  = qb   + (((size_t)b * NQ + qh) * S) * D;
  const u16* Kp = kbuf + (((size_t)b * NKV + kvh) * S) * D;
  const u16* Vp = vtb  + (((size_t)b * NKV + kvh) * D) * S;
  const int* am = amask + b * S;

  // Q A-fragments (d 0..31 and 32..63), loaded once
  const u16* qrow = Q + (size_t)(q0 + l15) * D + hf * 8;
  v16bf qa0 = mkfrag(ld4(qrow), ld4(qrow + 16));
  v16bf qa1 = mkfrag(ld4(qrow + 32), ld4(qrow + 48));

  v8f O[4];
  float mrow[8], lrow[8];
#pragma unroll
  for (int t = 0; t < 4; ++t) O[t] = vzero();
#pragma unroll
  for (int r = 0; r < 8; ++r) { mrow[r] = -1e30f; lrow[r] = 0.0f; }
  u16* pl = &lds[wave][0];
  const float scale = 0.125f;  // 1/sqrt(64)

  for (int kb0 = 0; kb0 <= q0 + 15; kb0 += 64) {
    // ---- load all K fragments for the 64-key tile in one clause ----
    uint4 klo[8], khi[8];
#pragma unroll
    for (int t = 0; t < 4; ++t) {
      const u16* kr = Kp + (size_t)(kb0 + t * 16 + l15) * D + hf * 8;
      klo[2 * t] = ld4(kr);           khi[2 * t] = ld4(kr + 16);       // d 0..31
      klo[2 * t + 1] = ld4(kr + 32);  khi[2 * t + 1] = ld4(kr + 48);   // d 32..63
    }
    int amv[4];
#pragma unroll
    for (int t = 0; t < 4; ++t) amv[t] = am[kb0 + t * 16 + l15];
    // ---- scores S = Q K^T over 64 keys (4 N-tiles x 2 K-steps) ----
    v8f st[4];
#pragma unroll
    for (int t = 0; t < 4; ++t) {
      v8f z = vzero();
      z = wmma_bf16(qa0, mkfrag(klo[2 * t], khi[2 * t]), z);
      z = wmma_bf16(qa1, mkfrag(klo[2 * t + 1], khi[2 * t + 1]), z);
      st[t] = z;
    }
    // ---- mask + scale, row max ----
    float mx[8];
#pragma unroll
    for (int r = 0; r < 8; ++r) mx[r] = -1e30f;
#pragma unroll
    for (int t = 0; t < 4; ++t) {
      const int kpos = kb0 + t * 16 + l15;
#pragma unroll
      for (int r = 0; r < 8; ++r) {
        const int qpos = q0 + r + 8 * hf;
        float v = st[t][r] * scale;
        v = (kpos <= qpos && amv[t] != 0) ? v : -1e30f;
        st[t][r] = v;
        mx[r] = fmaxf(mx[r], v);
      }
    }
#pragma unroll
    for (int msk = 1; msk < 16; msk <<= 1)
#pragma unroll
      for (int r = 0; r < 8; ++r) mx[r] = fmaxf(mx[r], __shfl_xor(mx[r], msk, 32));
    // ---- online softmax update ----
    float alpha[8], rs[8];
#pragma unroll
    for (int r = 0; r < 8; ++r) {
      float mn = fmaxf(mrow[r], mx[r]);
      alpha[r] = __expf(mrow[r] - mn);
      mrow[r] = mn;
      rs[r] = 0.0f;
    }
#pragma unroll
    for (int t = 0; t < 4; ++t)
#pragma unroll
      for (int r = 0; r < 8; ++r) {
        float p = __expf(st[t][r] - mrow[r]);
        rs[r] += p;
        pl[(r + 8 * hf) * 64 + t * 16 + l15] = f2bf(p);  // stage P in LDS
      }
#pragma unroll
    for (int msk = 1; msk < 16; msk <<= 1)
#pragma unroll
      for (int r = 0; r < 8; ++r) rs[r] += __shfl_xor(rs[r], msk, 32);
#pragma unroll
    for (int r = 0; r < 8; ++r) lrow[r] = lrow[r] * alpha[r] + rs[r];

    // intra-wave LDS exchange: lanes run in lockstep, just drain DS counter
    asm volatile("s_wait_dscnt 0x0" ::: "memory");

    // ---- load all V fragments for the tile in one clause ----
    uint4 vlo[8], vhi[8];
#pragma unroll
    for (int t = 0; t < 4; ++t) {
      const u16* vr = Vp + (size_t)(t * 16 + l15) * S + kb0 + hf * 8;
      vlo[2 * t] = ld4(vr);           vhi[2 * t] = ld4(vr + 16);       // kpos 0..31
      vlo[2 * t + 1] = ld4(vr + 32);  vhi[2 * t + 1] = ld4(vr + 48);   // kpos 32..63
    }
    // ---- rescale O, then O += P V ----
#pragma unroll
    for (int t = 0; t < 4; ++t)
#pragma unroll
      for (int r = 0; r < 8; ++r) O[t][r] *= alpha[r];
    const u16* pr = pl + l15 * 64 + hf * 8;
    v16bf pa0 = mkfrag(ld4(pr), ld4(pr + 16));        // kpos 0..31 of tile
    v16bf pa1 = mkfrag(ld4(pr + 32), ld4(pr + 48));   // kpos 32..63
#pragma unroll
    for (int t = 0; t < 4; ++t) {
      O[t] = wmma_bf16(pa0, mkfrag(vlo[2 * t], vhi[2 * t]), O[t]);
      O[t] = wmma_bf16(pa1, mkfrag(vlo[2 * t + 1], vhi[2 * t + 1]), O[t]);
    }
  }
  // ---- epilogue: normalize and store to [BS,H] bf16 for the output GEMM ----
  float inv[8];
#pragma unroll
  for (int r = 0; r < 8; ++r) inv[r] = lrow[r] > 0.0f ? 1.0f / lrow[r] : 0.0f;
#pragma unroll
  for (int t = 0; t < 4; ++t)
#pragma unroll
    for (int r = 0; r < 8; ++r) {
      float o = O[t][r] * inv[r];
      ab[(size_t)(b * S + q0 + r + 8 * hf) * H + qh * D + t * 16 + l15] = f2bf(o);
    }
}

// ---------------- host launch ----------------
extern "C" void kernel_launch(void* const* d_in, const int* in_sizes, int n_in,
                              void* d_out, int out_size, void* d_ws, size_t ws_size,
                              hipStream_t stream) {
  (void)in_sizes; (void)n_in; (void)out_size; (void)ws_size;
  const float* hs   = (const float*)d_in[0];
  const int* amask  = (const int*)d_in[1];
  const int* pos    = (const int*)d_in[2];
  const float* wq   = (const float*)d_in[3];
  const float* wk   = (const float*)d_in[4];
  const float* wv   = (const float*)d_in[5];
  const float* wo   = (const float*)d_in[6];
  char* ws = (char*)d_ws;
  u16* hsb  = (u16*)(ws + OFF_HSB);
  u16* wqT  = (u16*)(ws + OFF_WQT);
  u16* wkT  = (u16*)(ws + OFF_WKT);
  u16* wvT  = (u16*)(ws + OFF_WVT);
  u16* woT  = (u16*)(ws + OFF_WOT);
  float* qf = (float*)(ws + OFF_QF);
  float* kf = (float*)(ws + OFF_KF);
  float* vf = (float*)(ws + OFF_VF);
  u16* qbuf = (u16*)(ws + OFF_QB);
  u16* kbuf = (u16*)(ws + OFF_KB);
  u16* vtb  = (u16*)(ws + OFF_VT);
  u16* ab   = (u16*)(ws + OFF_AB);
  const int thr = 256;

  k_cvt<<<(BS * H + thr - 1) / thr, thr, 0, stream>>>(hs, hsb, BS * H);
  k_wtrans<<<(H * H + thr - 1) / thr, thr, 0, stream>>>(wq, wqT, H, H);
  k_wtrans<<<(H * HKV + thr - 1) / thr, thr, 0, stream>>>(wk, wkT, H, HKV);
  k_wtrans<<<(H * HKV + thr - 1) / thr, thr, 0, stream>>>(wv, wvT, H, HKV);
  k_wtrans<<<(H * H + thr - 1) / thr, thr, 0, stream>>>(wo, woT, H, H);

  k_gemm<<<dim3(H / 128, BS / 128), 256, 0, stream>>>(hsb, wqT, qf, BS, H, H);
  k_gemm<<<dim3(HKV / 128, BS / 128), 256, 0, stream>>>(hsb, wkT, kf, BS, HKV, H);
  k_gemm<<<dim3(HKV / 128, BS / 128), 256, 0, stream>>>(hsb, wvT, vf, BS, HKV, H);

  k_rope<<<(B * NQ * S * (D / 2) + thr - 1) / thr, thr, 0, stream>>>(qf, pos, qbuf, NQ);
  k_rope<<<(B * NKV * S * (D / 2) + thr - 1) / thr, thr, 0, stream>>>(kf, pos, kbuf, NKV);
  k_vtrans<<<(B * NKV * S * D + thr - 1) / thr, thr, 0, stream>>>(vf, vtb);

  k_flash<<<dim3(S / 64, NQ, B), 128, 0, stream>>>(qbuf, kbuf, vtb, amask, ab);

  k_gemm<<<dim3(H / 128, BS / 128), 256, 0, stream>>>(ab, woT, (float*)d_out, BS, H, H);
}
